// LinearSystem_57672820850886
// MI455X (gfx1250) — matrix operand order repaired and checked by reading, not compiled, not run on hardware
//
#include <hip/hip_runtime.h>
#include <hip/hip_bf16.h>

// ---------------------------------------------------------------------------
// Linear state-space simulation, blocked-parallel-scan formulation for MI455X.
//   states[b,t] = states[b,t-1] @ F^T + inputs[b,t] @ Bmat^T + eps_W[b,t] @ sqW^T
//   obs[b,t]    = states[b,t] @ H^T + eps_V[b,t] @ sqV^T
// HBM-bound (~470MB, ~14 GFLOP). All GEMMs via v_wmma_f32_16x16x32_bf16.
// v5: async global->LDS (ASYNCcnt) double-buffered staging in the scan inner
// loop (global latency hidden behind each step's WMMA chain), bf16 A-fragment
// LDS layout everywhere, independent WMMA chains to fill hazard slots.
// ---------------------------------------------------------------------------

#define BATCH 64
#define TT    8192
#define SD    64
#define ID    32
#define OD    32
#define LCH   64              // chunk length (F powers kept: F^1..F^64)
#define NCH   (TT / LCH)      // 128 chunks

#if __has_builtin(__builtin_amdgcn_global_load_async_to_lds_b128) && \
    __has_builtin(__builtin_amdgcn_s_wait_asynccnt)
#define USE_ASYNC 1
#endif

typedef __attribute__((ext_vector_type(16))) __bf16 v16bf;
typedef __attribute__((ext_vector_type(8)))  __bf16 v8bf;
typedef __attribute__((ext_vector_type(4)))  __bf16 v4bf;
typedef __attribute__((ext_vector_type(8)))  float  v8f;

static __device__ __forceinline__ int lane_id() { return (int)(threadIdx.x & 31u); }

#ifdef USE_ASYNC
// async copy of 16 bytes global -> LDS, tracked by ASYNCcnt.
// Builtin signature (from clang diagnostic): (v4i global*, v4i local*, Imm, Imm)
typedef int v4i_gcc __attribute__((vector_size(16)));
static __device__ __forceinline__ void async_ld_b128(const float* g, float* l) {
  __builtin_amdgcn_global_load_async_to_lds_b128(
      (__attribute__((address_space(1))) v4i_gcc*)(void*)g,
      (__attribute__((address_space(3))) v4i_gcc*)(void*)l, 0, 0);
}
#endif

// --- A-fragment slot math (ISA 7.12.2, 16-bit A 16x32):
// lane = row + 16*h, slot i; k%32 maps as: h=(k>>3)&1, i=(k&7)+((k>>4)<<3).
// LDS A-layout buffer: bf16[(kt*32 + lane)*16 + i] (32B per lane per kt).

// A fragment from row-major f32 src (converting load; used off the hot path)
static __device__ __forceinline__ v16bf load_A(const float* src, int lda, int ktile) {
  const int lane = lane_id();
  const int row  = lane & 15;
  const int half = lane >> 4;
  v16bf a;
#pragma unroll
  for (int i = 0; i < 16; ++i) {
    const int k = ktile * 32 + half * 8 + i + ((i >= 8) ? 8 : 0);
    a[i] = (__bf16)src[row * lda + k];
  }
  return a;
}

// A fragment directly from bf16 A-layout LDS: 2x ds_load_b128, no cvt
static __device__ __forceinline__ v16bf load_A_bf(const __bf16* sA, int ktile) {
  return ((const v16bf*)sA)[ktile * 32 + lane_id()];
}

// B fragment (direct): 32x16 tile of row-major KxN matrix M.
static __device__ __forceinline__ v16bf load_Bd(const float* M, int ld, int ktile, int ntile) {
  const int lane = lane_id();
  const int n    = lane & 15;
  const int half = lane >> 4;
  v16bf b;
#pragma unroll
  for (int i = 0; i < 16; ++i) {
    const int k = ktile * 32 + half * 8 + i + ((i >= 8) ? 8 : 0);
    b[i] = (__bf16)M[k * ld + ntile * 16 + n];
  }
  return b;
}

// B fragment for X @ M^T: B[k][n] = M[n][k], M row-major N x K.
static __device__ __forceinline__ v16bf load_Bt(const float* M, int ld, int ktile, int ntile) {
  const int lane = lane_id();
  const int n    = lane & 15;
  const int half = lane >> 4;
  v16bf b;
#pragma unroll
  for (int i = 0; i < 16; ++i) {
    const int k = ktile * 32 + half * 8 + i + ((i >= 8) ? 8 : 0);
    b[i] = (__bf16)M[(ntile * 16 + n) * ld + k];
  }
  return b;
}

// C/D fragment: 16x16 f32, lane = column, VGPR r = row (r + 8*lane_half).
static __device__ __forceinline__ v8f load_C(const float* src, int ld) {
  const int lane = lane_id();
  const int n    = lane & 15;
  const int half = lane >> 4;
  v8f c;
#pragma unroll
  for (int r = 0; r < 8; ++r) c[r] = src[(r + 8 * half) * ld + n];
  return c;
}

static __device__ __forceinline__ void store_C(float* dst, int ld, v8f c) {
  const int lane = lane_id();
  const int n    = lane & 15;
  const int half = lane >> 4;
#pragma unroll
  for (int r = 0; r < 8; ++r) dst[(r + 8 * half) * ld + n] = c[r];
}

// Scatter a C fragment (cols = ntile's 16) into a bf16 A-layout LDS buffer.
static __device__ __forceinline__ void storeC_to_A(__bf16* sA, v8f c, int ntile) {
  const int lane = lane_id();
  const int n    = lane & 15;
  const int half = lane >> 4;
  const int k    = ntile * 16 + n;
  const int kt   = k >> 5;
  const int kk   = k & 31;
  const int h    = (kk >> 3) & 1;
  const int i    = (kk & 7) + ((kk >> 4) << 3);
#pragma unroll
  for (int r = 0; r < 8; ++r)
    sA[(kt * 32 + (r + 8 * half) + 16 * h) * 16 + i] = (__bf16)c[r];
}

static __device__ __forceinline__ v8f wmma_bf16(v16bf a, v16bf b, v8f c) {
  return __builtin_amdgcn_wmma_f32_16x16x32_bf16(false, a, false, b, (short)0, c,
                                                 false, false);
}

// ---------------------------------------------------------------------------
// Kernel 1: Fpow[j] = F^(j+1), j = 0..LCH-1 (64x64 row-major each).
// 4 waves; wave w owns output n-tile w. Ping-pong bf16 A-layout LDS buffers.
// ---------------------------------------------------------------------------
__global__ void k_fpow(const float* __restrict__ F, float* __restrict__ Fpow) {
  __shared__ __align__(32) __bf16 cur[2][4][2 * 32 * 16];  // [pp][m-strip][kt,lane,i]
  const int tid = (int)threadIdx.x;      // 0..127
  const int w   = tid >> 5;              // wave -> n-tile

  // stage F into A-layout (and copy F^1 to Fpow) : rows 64 x 64 floats
  for (int idx = tid; idx < 512; idx += 128) {     // 8 floats per idx
    const int row = idx >> 3, q = idx & 7;
    const float4* s = (const float4*)(F + row * SD) + q * 2;
    const float4 e0 = s[0], e1 = s[1];
    ((float4*)Fpow)[idx * 2]     = e0;
    ((float4*)Fpow)[idx * 2 + 1] = e1;
    const int strip = row >> 4, r16 = row & 15;
    const int k0 = q * 8, kt = k0 >> 5, kk = k0 & 31;
    const int h = (kk >> 3) & 1, i = (kk & 7) + ((kk >> 4) << 3);
    *(v8bf*)&cur[0][strip][(kt * 32 + r16 + 16 * h) * 16 + i] =
        (v8bf){(__bf16)e0.x, (__bf16)e0.y, (__bf16)e0.z, (__bf16)e0.w,
               (__bf16)e1.x, (__bf16)e1.y, (__bf16)e1.z, (__bf16)e1.w};
  }
  __syncthreads();

  const v16bf bF0 = load_Bd(F, SD, 0, w);
  const v16bf bF1 = load_Bd(F, SD, 1, w);

  int p = 0;
  for (int j = 1; j < LCH; ++j) {
    float* out = Fpow + (size_t)j * SD * SD;
#pragma unroll
    for (int m = 0; m < 4; ++m) {
      const v16bf a0 = load_A_bf(cur[p][m], 0);
      const v16bf a1 = load_A_bf(cur[p][m], 1);
      v8f c = {};
      c = wmma_bf16(a0, bF0, c);
      c = wmma_bf16(a1, bF1, c);
      store_C(out + (m * 16) * SD + w * 16, SD, c);
      storeC_to_A(cur[p ^ 1][m], c, w);
    }
    p ^= 1;
    __syncthreads();
  }
}

// ---------------------------------------------------------------------------
// Kernel 2: per-chunk local scan (zero init), drive GEMM fused.
// grid = (NCH, BATCH/16), block = 128 (4 waves; wave w owns n-tile w).
// Async double-buffered staging of inputs/eps_W: step j+1's rows fly into LDS
// (ASYNCcnt) while step j's WMMA chain runs. Conversion to bf16 A-layout is
// LDS->LDS, off the serial state chain.
// ---------------------------------------------------------------------------
__global__ void k_scan_local(const float* __restrict__ inputs,
                             const float* __restrict__ eps_W,
                             const float* __restrict__ F,
                             const float* __restrict__ Bmat,
                             const float* __restrict__ sqW,
                             float* __restrict__ states) {
  const int c   = (int)blockIdx.x;
  const int b0  = (int)blockIdx.y * 16;
  const int tid = (int)threadIdx.x;      // 0..127
  const int w   = tid >> 5;              // wave -> n-tile

  __shared__ __align__(32) __bf16 sS[2][2 * 32 * 16];  // state strip, ping-pong
  __shared__ __align__(32) __bf16 sIn[32 * 16];        // A-layout inputs (K=32)
  __shared__ __align__(32) __bf16 sEw[2 * 32 * 16];    // A-layout eps_W (K=64)
#ifdef USE_ASYNC
  __shared__ __align__(16) float rawI[2][16 * ID];     // async-staged raw rows
  __shared__ __align__(16) float rawE[2][16 * SD];
#endif

  for (int i = tid; i < 512; i += 128) ((unsigned int*)sS[0])[i] = 0u;

  // per-wave B fragments (only this wave's n-tile): 5 frags, no spills
  const v16bf bF0 = load_Bt(F,    SD, 0, w);
  const v16bf bF1 = load_Bt(F,    SD, 1, w);
  const v16bf bW0 = load_Bt(sqW,  SD, 0, w);
  const v16bf bW1 = load_Bt(sqW,  SD, 1, w);
  const v16bf bB  = load_Bt(Bmat, ID, 0, w);

  // staging slot precompute (row/q fixed per thread)
  const int row = tid >> 3;              // 0..15
  const int q   = tid & 7;               // 0..7
  const int kI  = q * 4;                 // inputs: 4 floats per thread
  const int hI  = (kI >> 3) & 1;
  const int iI  = (kI & 7) + ((kI >> 4) << 3);
  __bf16* dstI  = &sIn[(row + 16 * hI) * 16 + iI];
  const int kE  = q * 8;                 // eps_W: 8 floats per thread
  const int ktE = kE >> 5, kkE = kE & 31;
  const int hE  = (kkE >> 3) & 1;
  const int iE  = (kkE & 7) + ((kkE >> 4) << 3);
  __bf16* dstE  = &sEw[(ktE * 32 + (row + 16 * hE)) * 16 + iE];

  const int t0 = c * LCH;

#ifdef USE_ASYNC
  // prologue: async-fetch step 0's rows
  async_ld_b128(inputs + ((size_t)(b0 + row) * TT + t0) * ID + q * 4,
                &rawI[0][row * ID + q * 4]);
  async_ld_b128(eps_W + ((size_t)(b0 + row) * TT + t0) * SD + q * 8,
                &rawE[0][row * SD + q * 8]);
  async_ld_b128(eps_W + ((size_t)(b0 + row) * TT + t0) * SD + q * 8 + 4,
                &rawE[0][row * SD + q * 8 + 4]);
  __builtin_amdgcn_s_wait_asynccnt(0);
  __syncthreads();
#endif

  int p = 0;
  for (int j = 0; j < LCH; ++j) {
    const int t = t0 + j;
#ifdef USE_ASYNC
    const int pb = j & 1, nb = pb ^ 1;
    if (j + 1 < LCH) {                   // prefetch step j+1 under step j's math
      const int tn = t + 1;
      async_ld_b128(inputs + ((size_t)(b0 + row) * TT + tn) * ID + q * 4,
                    &rawI[nb][row * ID + q * 4]);
      async_ld_b128(eps_W + ((size_t)(b0 + row) * TT + tn) * SD + q * 8,
                    &rawE[nb][row * SD + q * 8]);
      async_ld_b128(eps_W + ((size_t)(b0 + row) * TT + tn) * SD + q * 8 + 4,
                    &rawE[nb][row * SD + q * 8 + 4]);
    }
    // convert staged raw rows (LDS) -> bf16 A-layout (LDS)
    {
      const float4 vi = *(const float4*)&rawI[pb][row * ID + q * 4];
      *(v4bf*)dstI = (v4bf){(__bf16)vi.x, (__bf16)vi.y, (__bf16)vi.z, (__bf16)vi.w};
      const float4 e0 = *(const float4*)&rawE[pb][row * SD + q * 8];
      const float4 e1 = *(const float4*)&rawE[pb][row * SD + q * 8 + 4];
      *(v8bf*)dstE = (v8bf){(__bf16)e0.x, (__bf16)e0.y, (__bf16)e0.z, (__bf16)e0.w,
                            (__bf16)e1.x, (__bf16)e1.y, (__bf16)e1.z, (__bf16)e1.w};
    }
#else
    // synchronous fallback: coalesced stage + single conversion
    {
      const float4 vi = *((const float4*)(inputs + ((size_t)(b0 + row) * TT + t) * ID) + q);
      *(v4bf*)dstI = (v4bf){(__bf16)vi.x, (__bf16)vi.y, (__bf16)vi.z, (__bf16)vi.w};
      const float4* se = (const float4*)(eps_W + ((size_t)(b0 + row) * TT + t) * SD) + q * 2;
      const float4 e0 = se[0], e1 = se[1];
      *(v8bf*)dstE = (v8bf){(__bf16)e0.x, (__bf16)e0.y, (__bf16)e0.z, (__bf16)e0.w,
                            (__bf16)e1.x, (__bf16)e1.y, (__bf16)e1.z, (__bf16)e1.w};
    }
#endif
    __syncthreads();   // staging done; previous step's sS[p] writes visible

    const v16bf aS0 = load_A_bf(sS[p], 0);
    const v16bf aS1 = load_A_bf(sS[p], 1);
    const v16bf aI  = load_A_bf(sIn,   0);
    const v16bf aE0 = load_A_bf(sEw,   0);
    const v16bf aE1 = load_A_bf(sEw,   1);

    // two independent accumulation chains -> scheduler fills hazard slots
    v8f ca = {}, cb = {};
    ca = wmma_bf16(aI,  bB,  ca);
    cb = wmma_bf16(aE0, bW0, cb);
    ca = wmma_bf16(aE1, bW1, ca);
    cb = wmma_bf16(aS0, bF0, cb);
    ca = wmma_bf16(aS1, bF1, ca);
    const v8f cacc = ca + cb;

    store_C(states + ((size_t)b0 * TT + t) * SD + w * 16, TT * SD, cacc);
    storeC_to_A(sS[p ^ 1], cacc, w);
    p ^= 1;
#ifdef USE_ASYNC
    __builtin_amdgcn_s_wait_asynccnt(0);   // next raw buffer landed
#endif
    __syncthreads();   // frags consumed + next state written before re-staging
  }
  // chunk-final local state == states[:, t0+LCH-1, :] (already in d_out)
}

// ---------------------------------------------------------------------------
// Kernel 3: serial chunk combine (1 block, 4 waves):
//   sInit[0] = state0 ; sInit[c+1] = sInit[c] @ (F^LCH)^T + localfinal_c
// localfinal_c is read from states[:, (c+1)*LCH-1, :] in d_out.
// ---------------------------------------------------------------------------
__global__ void k_chunk_prefix(const float* __restrict__ state0,
                               const float* __restrict__ Fpow,
                               const float* __restrict__ states,
                               float* __restrict__ sInit) {
  __shared__ __align__(32) __bf16 sA[2][4][2 * 32 * 16];  // [pp][m-strip] A-layout
  __shared__ __align__(16) float  sF[2][BATCH * SD];      // fp32 copy for sInit writes
  const int tid = (int)threadIdx.x;
  const int w   = tid >> 5;

  for (int idx = tid; idx < 512; idx += 128) {    // 8 floats per idx
    const int row = idx >> 3, q = idx & 7;
    const float4* s = (const float4*)(state0 + row * SD) + q * 2;
    const float4 e0 = s[0], e1 = s[1];
    ((float4*)sF[0])[idx * 2]     = e0;
    ((float4*)sF[0])[idx * 2 + 1] = e1;
    const int strip = row >> 4, r16 = row & 15;
    const int k0 = q * 8, kt = k0 >> 5, kk = k0 & 31;
    const int h = (kk >> 3) & 1, i = (kk & 7) + ((kk >> 4) << 3);
    *(v8bf*)&sA[0][strip][(kt * 32 + r16 + 16 * h) * 16 + i] =
        (v8bf){(__bf16)e0.x, (__bf16)e0.y, (__bf16)e0.z, (__bf16)e0.w,
               (__bf16)e1.x, (__bf16)e1.y, (__bf16)e1.z, (__bf16)e1.w};
  }
  __syncthreads();

  const float* FL = Fpow + (size_t)(LCH - 1) * SD * SD;   // F^LCH
  const v16bf bF0 = load_Bt(FL, SD, 0, w);
  const v16bf bF1 = load_Bt(FL, SD, 1, w);

  int p = 0;
  for (int c = 0; c < NCH; ++c) {
    float* out = sInit + (size_t)c * BATCH * SD;
    for (int i = tid; i < BATCH * SD / 4; i += 128)
      ((float4*)out)[i] = ((const float4*)sF[p])[i];
    const int tl = c * LCH + LCH - 1;                     // chunk-final time index
#pragma unroll
    for (int m = 0; m < 4; ++m) {
      const v16bf a0 = load_A_bf(sA[p][m], 0);
      const v16bf a1 = load_A_bf(sA[p][m], 1);
      v8f cc = load_C(states + ((size_t)(m * 16) * TT + tl) * SD + w * 16, TT * SD);
      cc = wmma_bf16(a0, bF0, cc);
      cc = wmma_bf16(a1, bF1, cc);
      store_C(sF[p ^ 1] + (m * 16) * SD + w * 16, SD, cc);
      storeC_to_A(sA[p ^ 1][m], cc, w);
    }
    p ^= 1;
    __syncthreads();
  }
}

// ---------------------------------------------------------------------------
// Kernel 4: per-timestep correction + fused observation projection.
// grid = TT, block = 128 (4 waves). For t in chunk c at offset j:
//   states[:,t] += sInit[c] @ (F^(j+1))^T         (in-place fix-up)
//   obs[:,t]     = states[:,t] @ H^T + eps_V[:,t] @ sqV^T
// ---------------------------------------------------------------------------
__global__ void k_fix_obs(const float* __restrict__ Fpow,
                          const float* __restrict__ sInit,
                          const float* __restrict__ H,
                          const float* __restrict__ sqV,
                          const float* __restrict__ eps_V,
                          float* __restrict__ states,
                          float* __restrict__ obs) {
  const int t   = (int)blockIdx.x;
  const int c   = t / LCH;
  const int j   = t % LCH;
  const int tid = (int)threadIdx.x;
  const int w   = tid >> 5;

  __shared__ __align__(32) __bf16 si_a[4][2 * 32 * 16];  // sInit[c], A-layout
  __shared__ __align__(32) __bf16 sc_a[4][2 * 32 * 16];  // corrected states
  __shared__ __align__(32) __bf16 ev_a[4][32 * 16];      // eps_V rows (K=32)

  const float* Fp = Fpow  + (size_t)j * SD * SD;         // F^(j+1)
  const float* si = sInit + (size_t)c * BATCH * SD;

  for (int idx = tid; idx < 512; idx += 128) {           // sInit: 8 floats/idx
    const int row = idx >> 3, q = idx & 7;
    const float4* s = (const float4*)(si + row * SD) + q * 2;
    const float4 e0 = s[0], e1 = s[1];
    const int strip = row >> 4, r16 = row & 15;
    const int k0 = q * 8, kt = k0 >> 5, kk = k0 & 31;
    const int h = (kk >> 3) & 1, i = (kk & 7) + ((kk >> 4) << 3);
    *(v8bf*)&si_a[strip][(kt * 32 + r16 + 16 * h) * 16 + i] =
        (v8bf){(__bf16)e0.x, (__bf16)e0.y, (__bf16)e0.z, (__bf16)e0.w,
               (__bf16)e1.x, (__bf16)e1.y, (__bf16)e1.z, (__bf16)e1.w};
  }
  for (int idx = tid; idx < 512; idx += 128) {           // eps_V: 4 floats/idx
    const int row = idx >> 3, q = idx & 7;
    const float4 v = *((const float4*)(eps_V + ((size_t)row * TT + t) * OD) + q);
    const int strip = row >> 4, r16 = row & 15;
    const int k0 = q * 4;
    const int h = (k0 >> 3) & 1, i = (k0 & 7) + ((k0 >> 4) << 3);
    *(v4bf*)&ev_a[strip][(r16 + 16 * h) * 16 + i] =
        (v4bf){(__bf16)v.x, (__bf16)v.y, (__bf16)v.z, (__bf16)v.w};
  }
  __syncthreads();

  // phase 1: states[:,t] += sInit[c] @ (F^(j+1))^T
  {
    const v16bf bF0 = load_Bt(Fp, SD, 0, w);
    const v16bf bF1 = load_Bt(Fp, SD, 1, w);
#pragma unroll
    for (int m = 0; m < 4; ++m) {
      const v16bf a0 = load_A_bf(si_a[m], 0);
      const v16bf a1 = load_A_bf(si_a[m], 1);
      float* sg = states + ((size_t)(m * 16) * TT + t) * SD;
      v8f cc = load_C(sg + w * 16, TT * SD);
      cc = wmma_bf16(a0, bF0, cc);
      cc = wmma_bf16(a1, bF1, cc);
      store_C(sg + w * 16, TT * SD, cc);
      storeC_to_A(sc_a[m], cc, w);
    }
  }
  __syncthreads();

  // phase 2: obs from corrected states (still in LDS, bf16 A-layout)
  {
    const int nt = w & 1;
    const int mh = w >> 1;
    const v16bf bH0 = load_Bt(H,   SD, 0, nt);
    const v16bf bH1 = load_Bt(H,   SD, 1, nt);
    const v16bf bV  = load_Bt(sqV, OD, 0, nt);
#pragma unroll
    for (int mm = 0; mm < 2; ++mm) {
      const int m = mh * 2 + mm;
      const v16bf a0 = load_A_bf(sc_a[m], 0);
      const v16bf a1 = load_A_bf(sc_a[m], 1);
      const v16bf aV = load_A_bf(ev_a[m], 0);
      v8f ca = {}, cb = {};
      ca = wmma_bf16(aV, bV,  ca);
      cb = wmma_bf16(a0, bH0, cb);
      ca = wmma_bf16(a1, bH1, ca);
      const v8f cc = ca + cb;
      store_C(obs + ((size_t)(m * 16) * TT + t) * OD + nt * 16, TT * OD, cc);
    }
  }
}

// ---------------------------------------------------------------------------
extern "C" void kernel_launch(void* const* d_in, const int* in_sizes, int n_in,
                              void* d_out, int out_size, void* d_ws, size_t ws_size,
                              hipStream_t stream) {
  (void)in_sizes; (void)n_in; (void)out_size; (void)ws_size;
  const float* state0 = (const float*)d_in[0];
  const float* inputs = (const float*)d_in[1];
  const float* F      = (const float*)d_in[2];
  const float* Bmat   = (const float*)d_in[3];
  const float* H      = (const float*)d_in[4];
  const float* sqW    = (const float*)d_in[5];
  const float* sqV    = (const float*)d_in[6];
  const float* eps_W  = (const float*)d_in[7];
  const float* eps_V  = (const float*)d_in[8];

  float* states = (float*)d_out;                          // [B,T,S]
  float* obs    = states + (size_t)BATCH * TT * SD;       // [B,T,O]

  float* ws    = (float*)d_ws;
  float* Fpow  = ws;                                      // LCH*SD*SD    (1 MB)
  float* sInit = Fpow + (size_t)LCH * SD * SD;            // NCH*BATCH*SD (2 MB)

  hipLaunchKernelGGL(k_fpow, dim3(1), dim3(128), 0, stream, F, Fpow);
  hipLaunchKernelGGL(k_scan_local, dim3(NCH, BATCH / 16), dim3(128), 0, stream,
                     inputs, eps_W, F, Bmat, sqW, states);
  hipLaunchKernelGGL(k_chunk_prefix, dim3(1), dim3(128), 0, stream,
                     state0, Fpow, states, sInit);
  hipLaunchKernelGGL(k_fix_obs, dim3(TT), dim3(128), 0, stream,
                     Fpow, sInit, H, sqV, eps_V, states, obs);
}
